// ScaledDotProductAttention_31035433681575
// MI455X (gfx1250) — matrix operand order repaired and checked by reading, not compiled, not run on hardware
//
#include <hip/hip_runtime.h>
#include <hip/hip_bf16.h>
#include <stdint.h>

// CDNA5 WMMA fragment types (wave32): 16x16x32 bf16 -> f32
typedef __attribute__((ext_vector_type(16))) __bf16 v16bf;
typedef __attribute__((ext_vector_type(8)))  float  v8f;
// TDM descriptor groups (amdgpu-toolchain clang-23: 6-arg builtin signature)
typedef __attribute__((ext_vector_type(4)))  unsigned int u32x4;
typedef __attribute__((ext_vector_type(8)))  int i32x8;
typedef __attribute__((ext_vector_type(4)))  int i32x4;

#define B_     32
#define LQ     2048
#define LK     2048
#define DD     64
#define QTILE  16
#define KCHUNK 128

static __device__ __forceinline__ unsigned ldsof(const void* p) {
    // generic pointer to static LDS: low 32 bits carry the LDS byte offset
    return (unsigned)(uintptr_t)p;
}

// Issue one TDM 2D tile DMA: global f32 tile (rows x rowlen, row-major,
// stride == rowlen) -> LDS at lds_off. Tensor == tile, so fully in-bounds.
// D# packing per CDNA5 ISA 8.3/8.4 (group0 128b, group1 256b; groups 2/3 zero).
static __device__ __forceinline__ void tdm_load_2d_f32(const float* gsrc,
                                                       unsigned lds_off,
                                                       int rows, int rowlen) {
    const unsigned long long ga = (unsigned long long)(uintptr_t)gsrc;
    u32x4 g0;
    g0[0] = 1u;                                              // count=1 valid descriptor
    g0[1] = lds_off;                                         // lds_addr[31:0]
    g0[2] = (unsigned)(ga & 0xFFFFFFFFu);                    // global_addr[31:0]
    g0[3] = (unsigned)((ga >> 32) & 0x1FFFFFFu) | (2u << 30); // ga[56:32] | type=2
    i32x8 g1;
    g1[0] = 2 << 16;                                         // wg_mask=0, data_size=4B
    g1[1] = (rowlen & 0xFFFF) << 16;                         // tensor_dim0[15:0]
    g1[2] = ((rowlen >> 16) & 0xFFFF) | ((rows & 0xFFFF) << 16);   // td0 hi | tensor_dim1 lo
    g1[3] = ((rows >> 16) & 0xFFFF) | ((rowlen & 0xFFFF) << 16);   // td1 hi | tile_dim0
    g1[4] = rows & 0xFFFF;                                   // tile_dim1 | tile_dim2=0
    g1[5] = rowlen;                                          // tensor_dim0_stride[31:0]
    g1[6] = 0;                                               // stride0 hi | stride1 lo
    g1[7] = 0;                                               // stride1 hi
    const i32x4 gz4 = {0, 0, 0, 0};                          // groups 2/3: dims 2..4 unused
    const i32x8 gz8 = {0, 0, 0, 0, 0, 0, 0, 0};              // extended words: unused
    __builtin_amdgcn_tensor_load_to_lds(g0, g1, gz4, gz4, gz8, 0);
}

__global__ __launch_bounds__(256)
void sdpa_wmma_kernel(const float* __restrict__ qg,
                      const float* __restrict__ kg,
                      const float* __restrict__ vg,
                      const int*   __restrict__ maskg,
                      float* __restrict__ outg,
                      float* __restrict__ attng)
{
    // LDS: 128K scores + 64K bf16 probs + 2K Q + 64K f32 K/V double-buffer + 4K reduce
    //    = ~262KB (< 320KB/WGP)
    __shared__ float  S[QTILE][LK];
    __shared__ __bf16 Pb[QTILE][LK];
    __shared__ __bf16 Qb[QTILE][DD];
    __shared__ float  KVf[2][KCHUNK][DD];
    __shared__ float  Cred[4][16][16];

    const int b    = blockIdx.y;
    const int q0   = blockIdx.x * QTILE;
    const int tid  = threadIdx.x;
    const int lane = tid & 31;
    const int wave = tid >> 5;
    const int mrow = lane & 15;   // A row / B,C,D column owned by this lane
    const int kh   = lane >> 4;   // half-wave index

    // ---- prime TDM: K chunk 0, and load Q tile (16x64) as bf16 ----
    if (wave == 0)
        tdm_load_2d_f32(&kg[((size_t)b * LK) * DD], ldsof(&KVf[0][0][0]), KCHUNK, DD);
    for (int idx = tid; idx < QTILE * DD; idx += 256) {
        const int r = idx >> 6, c = idx & 63;
        Qb[r][c] = (__bf16)qg[((size_t)b * LQ + q0 + r) * DD + c];   // hw cvt
    }

    // ================= Phase 1: S = (Q @ K^T) / 8 =================
    for (int kc = 0; kc < LK; kc += KCHUNK) {
        const int buf = (kc / KCHUNK) & 1;
        if (kc + KCHUNK < LK) {               // prefetch next chunk via TDM
            if (wave == 0) {
                tdm_load_2d_f32(&kg[((size_t)b * LK + kc + KCHUNK) * DD],
                                ldsof(&KVf[buf ^ 1][0][0]), KCHUNK, DD);
                __builtin_amdgcn_s_wait_tensorcnt((short)1);  // current chunk landed
            }
        } else {
            if (wave == 0)
                __builtin_amdgcn_s_wait_tensorcnt((short)0);
        }
        __syncthreads();

        const int n0 = wave * 16;             // this wave's 16 key columns in the chunk
        v8f c = {};
        #pragma unroll
        for (int ks = 0; ks < 2; ++ks) {      // K-dim (d=64) in two 32-slices
            v16bf a, bb;
            #pragma unroll
            for (int e = 0; e < 16; ++e) {    // A 16x32 bf16 layout (ISA 7.12.2)
                const int vv = e >> 1, pos = e & 1;
                const int kk = ((vv < 4) ? 0 : 16) + (vv & 3) * 2 + pos + kh * 8 + ks * 32;
                a[e] = Qb[mrow][kk];
            }
            #pragma unroll
            for (int e = 0; e < 16; ++e) {    // B 32x16: lane=column, K split per half-wave
                const int kk = ((e < 8) ? e : e + 8) + kh * 8 + ks * 32;
                bb[e] = (__bf16)KVf[buf][n0 + mrow][kk];   // B[kk][n] = K[n0+n][kk], hw cvt
            }
            c = __builtin_amdgcn_wmma_f32_16x16x32_bf16(false, a, false, bb,
                                                        (short)0, c, false, false);
        }
        #pragma unroll
        for (int vv = 0; vv < 8; ++vv)        // C/D: M = vv + 8*kh, N = mrow
            S[vv + 8 * kh][kc + n0 + mrow] = c[vv] * 0.125f;  // /sqrt(64)
        __syncthreads();
    }

    // ---- prefetch V chunk 0 via TDM; it overlaps the whole softmax phase ----
    if (wave == 0)
        tdm_load_2d_f32(&vg[((size_t)b * LK) * DD], ldsof(&KVf[0][0][0]), KCHUNK, DD);

    // ================= Phase 2: masked softmax per row =================
    // reference: z=a*m; p=softmax(z); p*=m; attn=p/(sum(p)+eps)
    //         => attn_j = m_j*exp(a_j - zmax) / (sum_{m=1} exp(a - zmax) + eps)
    for (int rr = 0; rr < 2; ++rr) {
        const int r = wave * 2 + rr;
        const size_t mbase = ((size_t)b * LQ + q0 + r) * (size_t)LK;

        uint64_t mbits = 0;                   // cache 64 mask bits/lane -> read mask once
        float zmax = -3.0e38f;
        #pragma unroll 1
        for (int i = 0; i < LK / 32; ++i) {
            const int j = lane + 32 * i;
            const int m = maskg[mbase + j];
            mbits |= ((uint64_t)(m & 1)) << i;
            const float s = S[r][j];
            zmax = fmaxf(zmax, m ? s : 0.0f);
        }
        #pragma unroll
        for (int off = 16; off >= 1; off >>= 1)
            zmax = fmaxf(zmax, __shfl_xor(zmax, off, 32));

        float sum = 0.0f;
        #pragma unroll 1
        for (int i = 0; i < LK / 32; ++i) {
            const int j = lane + 32 * i;
            const float s = S[r][j];
            const float p = ((mbits >> i) & 1) ? __expf(s - zmax) : 0.0f;
            S[r][j] = p;
            sum += p;
        }
        #pragma unroll
        for (int off = 16; off >= 1; off >>= 1)
            sum += __shfl_xor(sum, off, 32);

        const float inv = 1.0f / (sum + 1e-13f);
        #pragma unroll 1
        for (int i = 0; i < LK / 32; ++i) {
            const int j = lane + 32 * i;
            const float p = S[r][j] * inv;
            attng[mbase + j] = p;             // coalesced attn_out stream
            Pb[r][j] = (__bf16)p;             // bf16 probs for P @ V (hw cvt)
        }
    }
    __syncthreads();

    // ================= Phase 3: out = P @ V =================
    const int t    = wave & 3;      // output column tile: cols t*16 .. t*16+15
    const int kpar = wave >> 2;     // K-parallel half (0/1), reduced at the end
    v8f cacc = {};
    for (int kc = 0; kc < LK; kc += KCHUNK) {
        const int buf = (kc / KCHUNK) & 1;
        if (kc + KCHUNK < LK) {               // prefetch next V chunk via TDM
            if (wave == 0) {
                tdm_load_2d_f32(&vg[((size_t)b * LK + kc + KCHUNK) * DD],
                                ldsof(&KVf[buf ^ 1][0][0]), KCHUNK, DD);
                __builtin_amdgcn_s_wait_tensorcnt((short)1);
            }
        } else {
            if (wave == 0)
                __builtin_amdgcn_s_wait_tensorcnt((short)0);
        }
        __syncthreads();

        #pragma unroll
        for (int s4 = 0; s4 < 2; ++s4) {
            const int kloc = (2 * s4 + kpar) * 32;   // interleaved 32-wide K steps
            v16bf a, bb;
            #pragma unroll
            for (int e = 0; e < 16; ++e) {
                const int vv = e >> 1, pos = e & 1;
                const int kk = ((vv < 4) ? 0 : 16) + (vv & 3) * 2 + pos + kh * 8;
                a[e] = Pb[mrow][kc + kloc + kk];
            }
            #pragma unroll
            for (int e = 0; e < 16; ++e) {
                const int kk = ((e < 8) ? e : e + 8) + kh * 8;
                bb[e] = (__bf16)KVf[buf][kloc + kk][t * 16 + mrow];  // B[kk][n]=V[kk][t*16+n]
            }
            cacc = __builtin_amdgcn_wmma_f32_16x16x32_bf16(false, a, false, bb,
                                                           (short)0, cacc, false, false);
        }
        __syncthreads();
    }
    if (kpar == 1) {
        #pragma unroll
        for (int vv = 0; vv < 8; ++vv)
            Cred[t][vv + 8 * kh][mrow] = cacc[vv];
    }
    __syncthreads();
    if (kpar == 0) {
        #pragma unroll
        for (int vv = 0; vv < 8; ++vv) {
            const int M = vv + 8 * kh;
            const float val = cacc[vv] + Cred[t][M][mrow];
            outg[((size_t)b * LQ + q0 + M) * DD + t * 16 + mrow] = val;
        }
    }
}

extern "C" void kernel_launch(void* const* d_in, const int* in_sizes, int n_in,
                              void* d_out, int out_size, void* d_ws, size_t ws_size,
                              hipStream_t stream) {
    (void)in_sizes; (void)n_in; (void)out_size; (void)d_ws; (void)ws_size;
    const float* q    = (const float*)d_in[0];
    const float* k    = (const float*)d_in[1];
    const float* v    = (const float*)d_in[2];
    const int*   mask = (const int*)d_in[3];
    float* out  = (float*)d_out;                      // [B, Lq, 64] first
    float* attn = out + (size_t)B_ * LQ * DD;         // then [B, Lq, Lk]
    dim3 grid(LQ / QTILE, B_);
    sdpa_wmma_kernel<<<grid, 256, 0, stream>>>(q, k, v, mask, out, attn);
}
